// DiTBlock_7206955123294
// MI455X (gfx1250) — matrix operand order repaired and checked by reading, not compiled, not run on hardware
//
#include <hip/hip_runtime.h>

typedef _Float16 f16;
typedef __attribute__((ext_vector_type(16))) _Float16 v16h;
typedef __attribute__((ext_vector_type(8)))  _Float16 h8;
typedef __attribute__((ext_vector_type(8)))  float    v8f;

#define HID   768
#define ADA   (6*HID)        // 4608
#define BATCH 64
#define SEQ   1024
#define NROW  (BATCH*SEQ)    // 65536
#define HEADS 12
#define HDIM  64

// 1/sqrt(64) * log2(e): folded into Q so attention uses exp2 directly
#define QSCALE 0.18033688011112042f

union FragU { v16h h; unsigned int u[8]; };

static __device__ __forceinline__ v8f wmma_f16(v16h a, v16h b, v8f c) {
  // v_wmma_f32_16x16x32_f16  D = A(16x32) * B(32x16) + C
  return __builtin_amdgcn_wmma_f32_16x16x32_f16(false, a, false, b, (short)0, c,
                                                false, false);
}
// 16-bit A-matrix 16x32 per-lane K index: lanes 0-15 hold K {0..7,16..23},
// lanes 16-31 hold K {8..15,24..31}; VGPR v packs pair (k, k+1).
static __device__ __forceinline__ int a_k(int lhalf, int v) {
  return lhalf * 8 + (v < 4 ? 2 * v : 16 + 2 * (v - 4));
}
// 16-bit B-matrix 32x16 per-lane K index: lanes 0-15 K {0..7,16..23}, lanes 16-31 K {8..15,24..31}.
static __device__ __forceinline__ int b_k(int lhalf, int v) {
  return lhalf * 8 + (v < 4 ? 2 * v : 16 + 2 * (v - 4));
}

// Async global -> LDS copy, 16B per lane (tracked by ASYNCcnt, not LOADcnt).
// LDS offset = low 32 bits of the flat shared address (aperture in [63:32]).
static __device__ __forceinline__ void async_ld_b128(void* lds, const void* g) {
  unsigned loff = (unsigned)(size_t)lds;
  asm volatile("global_load_async_to_lds_b128 %0, %1, off"
               :: "v"(loff), "v"(g) : "memory");
}
static __device__ __forceinline__ void async_wait0() {
  asm volatile("s_wait_asynccnt 0" ::: "memory");
}

// xor-shuffle within a 16-lane group via v_permlane16_b32 (pure VALU, no LDS)
static __device__ __forceinline__ float pl16(float x, unsigned s0, unsigned s1) {
  union { float f; unsigned u; } a, r;
  a.f = x;
  r.u = __builtin_amdgcn_permlane16(a.u, a.u, s0, s1, false, false);
  return r.f;
}
static __device__ __forceinline__ float rowmax16(float x) {
  x = fmaxf(x, pl16(x, 0x67452301u, 0xEFCDAB89u));  // xor 1
  x = fmaxf(x, pl16(x, 0x45670123u, 0xCDEF89ABu));  // xor 3
  x = fmaxf(x, pl16(x, 0x32107654u, 0xBA98FEDCu));  // xor 4
  x = fmaxf(x, pl16(x, 0xFEDCBA98u, 0x76543210u));  // xor 8
  return x;
}

// ---------------------------------------------------------------------------
// adaLN: c[b, 0:4608] = silu(cond[b]) @ w_ada + b_ada     (tiny GEMM)
// ---------------------------------------------------------------------------
__global__ __launch_bounds__(256) void adaln_kernel(
    const float* __restrict__ cond, const float* __restrict__ w,
    const float* __restrict__ bb, float* __restrict__ cout) {
  __shared__ float sc[HID];
  const int b = blockIdx.x;
  const int tid = threadIdx.x;
  for (int i = tid; i < HID; i += 256) {
    float v = cond[b * HID + i];
    sc[i] = v / (1.0f + __expf(-v));
  }
  __syncthreads();
  const int j = blockIdx.y * 256 + tid;
  float acc = bb[j];
  for (int k = 0; k < HID; ++k) acc += sc[k] * w[(size_t)k * ADA + j];
  cout[(size_t)b * ADA + j] = acc;
}

// ---------------------------------------------------------------------------
// f32 -> f16 conversion (weights)
// ---------------------------------------------------------------------------
__global__ __launch_bounds__(256) void cvt_f16_kernel(
    const float* __restrict__ in, f16* __restrict__ out, int n) {
  int i = blockIdx.x * 256 + threadIdx.x;
  if (i < n) out[i] = (f16)in[i];
}

// ---------------------------------------------------------------------------
// LayerNorm + adaLN modulate -> f16. One wave per row of 768. (HBM-bound)
// ---------------------------------------------------------------------------
__global__ __launch_bounds__(256) void ln_mod_kernel(
    const float* __restrict__ x, const float* __restrict__ c, int shift_off,
    int scale_off, const float* __restrict__ lns, const float* __restrict__ lnb,
    f16* __restrict__ out) {
  const int lane = threadIdx.x & 31;
  const int wave = threadIdx.x >> 5;
  const int row = blockIdx.x * 8 + wave;
  const float* xr = x + (size_t)row * HID;
  float vals[24];
  float s = 0.f, ss = 0.f;
#pragma unroll
  for (int j = 0; j < 24; ++j) {
    float v = xr[lane + j * 32];
    vals[j] = v; s += v; ss += v * v;
  }
#pragma unroll
  for (int m = 1; m < 32; m <<= 1) { s += __shfl_xor(s, m); ss += __shfl_xor(ss, m); }
  const float mu = s * (1.0f / HID);
  const float var = ss * (1.0f / HID) - mu * mu;
  const float rs = rsqrtf(var + 1e-6f);
  const int b = row >> 10;
  const float* cb = c + (size_t)b * ADA;
  f16* orow = out + (size_t)row * HID;
#pragma unroll
  for (int j = 0; j < 24; ++j) {
    int n = lane + j * 32;
    float y = (vals[j] - mu) * rs * lns[n] + lnb[n];
    y = y * (1.0f + cb[scale_off + n]) + cb[shift_off + n];
    orow[n] = (f16)y;
  }
}

// ---------------------------------------------------------------------------
// WMMA GEMM: C[M,N] = A[M,K](f16) * W[K,N](f16) + bias, fused epilogues.
// Block tile 128x128x32, 256 threads = 8 waves in 2(m) x 4(n); each wave does
// 64x32 = 4x2 WMMA tiles. Double-buffered LDS, one barrier per k-step.
// A tile is staged with GLOBAL_LOAD_ASYNC_TO_LDS_B128 (ASYNCcnt pipe);
// B tile needs a transpose so it stays on the register path.
// EPI: 0=store f16, 1=silu f16, 2=resid+gate f32, 3=scaled f16 (Q).
// ---------------------------------------------------------------------------
template <int EPI>
__global__ __launch_bounds__(256) void gemm_wmma_kernel(
    const f16* __restrict__ A, const f16* __restrict__ W,
    const float* __restrict__ bias, void* __restrict__ out,
    const float* __restrict__ resid, const float* __restrict__ gate,
    int M, int N, int K) {
  __shared__ f16 sA[2][128][32];   // A tile, row-major
  __shared__ f16 sBt[2][128][32];  // B tile transposed: sBt[.][n][k]
  const int tid = threadIdx.x;
  const int lane = tid & 31, wave = tid >> 5;
  const int wm = wave >> 2, wn = wave & 3;
  const int lhalf = lane >> 4, l15 = lane & 15;
  const int bm = blockIdx.x * 128, bn = blockIdx.y * 128;

  v8f acc[4][2];
#pragma unroll
  for (int i = 0; i < 4; ++i)
#pragma unroll
    for (int j = 0; j < 2; ++j) acc[i][j] = (v8f)0.0f;

  // per-thread staging coordinates
  const int arow = tid >> 1, acol = (tid & 1) * 16;
  const f16* aptr = A + (size_t)(bm + arow) * K + acol;
  const int bk = tid >> 3, bn2 = (tid & 7) * 16;
  const f16* bptr = W + (size_t)bk * N + bn + bn2;

  h8 rb0, rb1;
  // preload k-step 0
  async_ld_b128(&sA[0][arow][acol], aptr);
  async_ld_b128(&sA[0][arow][acol + 8], aptr + 8);
  rb0 = *(const h8*)(bptr);
  rb1 = *(const h8*)(bptr + 8);
#pragma unroll
  for (int i = 0; i < 8; ++i) {
    sBt[0][bn2 + i][bk]     = rb0[i];
    sBt[0][bn2 + 8 + i][bk] = rb1[i];
  }
  async_wait0();
  __syncthreads();

  const int KT = K / 32;
  for (int kt = 0; kt < KT; ++kt) {
    const int cur = kt & 1;
    const int nxt = cur ^ 1;
    if (kt + 1 < KT) {  // prefetch next tile (async A straight into LDS)
      const f16* ap = aptr + (kt + 1) * 32;
      async_ld_b128(&sA[nxt][arow][acol], ap);
      async_ld_b128(&sA[nxt][arow][acol + 8], ap + 8);
      const f16* bp = bptr + (size_t)(kt + 1) * 32 * N;
      rb0 = *(const h8*)bp;
      rb1 = *(const h8*)(bp + 8);
    }

    FragU af[4];
#pragma unroll
    for (int i = 0; i < 4; ++i) {
      int m = wm * 64 + i * 16 + l15;
#pragma unroll
      for (int v = 0; v < 8; ++v)
        af[i].u[v] = *(const unsigned int*)&sA[cur][m][a_k(lhalf, v)];
    }
    FragU bf[2];
#pragma unroll
    for (int j = 0; j < 2; ++j) {
      int n = wn * 32 + j * 16 + l15;
#pragma unroll
      for (int v = 0; v < 8; ++v)
        bf[j].u[v] = *(const unsigned int*)&sBt[cur][n][b_k(lhalf, v)];
    }
#pragma unroll
    for (int i = 0; i < 4; ++i)
#pragma unroll
      for (int j = 0; j < 2; ++j)
        acc[i][j] = wmma_f16(af[i].h, bf[j].h, acc[i][j]);

    if (kt + 1 < KT) {
#pragma unroll
      for (int i = 0; i < 8; ++i) {
        sBt[nxt][bn2 + i][bk]     = rb0[i];
        sBt[nxt][bn2 + 8 + i][bk] = rb1[i];
      }
    }
    async_wait0();
    __syncthreads();
  }

  // epilogue (C layout: m = 8*lhalf + r, n = l15)
#pragma unroll
  for (int i = 0; i < 4; ++i)
#pragma unroll
    for (int j = 0; j < 2; ++j) {
      const int col = bn + wn * 32 + j * 16 + l15;
      const float bv = bias[col];
#pragma unroll
      for (int r = 0; r < 8; ++r) {
        const int row = bm + wm * 64 + i * 16 + lhalf * 8 + r;
        const float v = acc[i][j][r] + bv;
        const size_t idx = (size_t)row * N + col;
        if (EPI == 0) {
          ((f16*)out)[idx] = (f16)v;
        } else if (EPI == 1) {
          ((f16*)out)[idx] = (f16)(v / (1.0f + __expf(-v)));
        } else if (EPI == 3) {
          ((f16*)out)[idx] = (f16)(v * QSCALE);
        } else {
          const int b = row >> 10;
          ((float*)out)[idx] = resid[idx] + gate[(size_t)b * ADA + col] * v;
        }
      }
    }
}

// ---------------------------------------------------------------------------
// Flash attention: Q,K,V,O are [B, L, HEADS*HDIM] f16; Q pre-scaled so logits
// are already in log2 domain. One block per (b, h, 128-query slab); each wave
// owns 16 queries, streams 64-key blocks. K tile staged via async-to-LDS;
// row-sum of P accumulated by an extra WMMA against an all-ones B fragment.
// ---------------------------------------------------------------------------
__global__ __launch_bounds__(256) void flash_attn_kernel(
    const f16* __restrict__ Q, const f16* __restrict__ Kd,
    const f16* __restrict__ V, f16* __restrict__ O) {
  const int bid = blockIdx.x;
  const int qblk = bid & 7;          // L/128 = 8
  const int bh = bid >> 3;
  const int h = bh % HEADS;
  const int b = bh / HEADS;

  __shared__ f16 sK[64][64];         // [key][feat]
  __shared__ f16 sVt[64][64];        // transposed: [feat][key]
  __shared__ f16 sP[8][16][32];      // per-wave P repack scratch

  const int tid = threadIdx.x, lane = tid & 31, wave = tid >> 5;
  const int lhalf = lane >> 4, l15 = lane & 15;
  const int qbase = qblk * 128 + wave * 16;

  // all-ones B fragment (32x16 of 1.0h): row-sum generator
  FragU onesf;
#pragma unroll
  for (int v = 0; v < 8; ++v) onesf.u[v] = 0x3C003C00u;

  // Q fragments (16x64 split into two 16x32 A-frags), loaded straight from HBM
  FragU qf[2];
  {
    const f16* qrow = Q + ((size_t)(b * SEQ + qbase + l15)) * HID + h * HDIM;
#pragma unroll
    for (int f = 0; f < 2; ++f)
#pragma unroll
      for (int v = 0; v < 8; ++v)
        qf[f].u[v] = *(const unsigned int*)(qrow + f * 32 + a_k(lhalf, v));
  }

  float mrun[8];
  v8f ot[5];                         // ot[0..3] = O accum, ot[4] = row sums
#pragma unroll
  for (int r = 0; r < 8; ++r) mrun[r] = -3.0e38f;
#pragma unroll
  for (int j = 0; j < 5; ++j) ot[j] = (v8f)0.0f;

  for (int kb0 = 0; kb0 < SEQ; kb0 += 64) {
    __syncthreads();
    {  // stage 64 keys: K async straight to LDS, V transposed via registers
      int r = tid >> 2, cs = (tid & 3) * 16;
      const f16* ks = Kd + ((size_t)(b * SEQ + kb0 + r)) * HID + h * HDIM + cs;
      const f16* vs = V  + ((size_t)(b * SEQ + kb0 + r)) * HID + h * HDIM + cs;
      async_ld_b128(&sK[r][cs], ks);
      async_ld_b128(&sK[r][cs + 8], ks + 8);
      h8 v0 = *(const h8*)vs;
      h8 v1 = *(const h8*)(vs + 8);
#pragma unroll
      for (int i = 0; i < 8; ++i) {
        sVt[cs + i][r]     = v0[i];
        sVt[cs + 8 + i][r] = v1[i];
      }
    }
    async_wait0();
    __syncthreads();

#pragma unroll
    for (int s = 0; s < 2; ++s) {  // two 32-key halves
      // S = Q @ K^T (already log2-scaled): two 16x16 tiles
      v8f st[2];
      st[0] = (v8f)0.0f; st[1] = (v8f)0.0f;
#pragma unroll
      for (int t = 0; t < 2; ++t) {
        const int key = s * 32 + t * 16 + l15;
#pragma unroll
        for (int f = 0; f < 2; ++f) {
          FragU kf;
#pragma unroll
          for (int v = 0; v < 8; ++v)
            kf.u[v] = *(const unsigned int*)&sK[key][f * 32 + b_k(lhalf, v)];
          st[t] = wmma_f16(qf[f].h, kf.h, st[t]);
        }
      }
      // online softmax over these 32 keys; row r lives in one half-wave
#pragma unroll
      for (int r = 0; r < 8; ++r) {
        const float s0 = st[0][r], s1 = st[1][r];
        const float mx = rowmax16(fmaxf(s0, s1));      // VALU permlane reduce
        const float mn = fmaxf(mrun[r], mx);
        const float alpha = exp2f(mrun[r] - mn);       // v_exp_f32 is 2^x
        const float p0 = exp2f(s0 - mn), p1 = exp2f(s1 - mn);
        mrun[r] = mn;
#pragma unroll
        for (int j = 0; j < 5; ++j) ot[j][r] *= alpha; // includes row-sum acc
        const int m_loc = lhalf * 8 + r;
        sP[wave][m_loc][l15]      = (f16)p0;
        sP[wave][m_loc][16 + l15] = (f16)p1;
      }
      asm volatile("s_wait_dscnt 0" ::: "memory");  // P LDS RAW within wave

      // O += P(16x32) @ V(32x64);  row-sum += P @ ones
      FragU pf;
#pragma unroll
      for (int v = 0; v < 8; ++v)
        pf.u[v] = *(const unsigned int*)&sP[wave][l15][a_k(lhalf, v)];
#pragma unroll
      for (int j = 0; j < 4; ++j) {
        FragU vf;
        const int feat = j * 16 + l15;
#pragma unroll
        for (int v = 0; v < 8; ++v)
          vf.u[v] = *(const unsigned int*)&sVt[feat][s * 32 + b_k(lhalf, v)];
        ot[j] = wmma_f16(pf.h, vf.h, ot[j]);
      }
      ot[4] = wmma_f16(pf.h, onesf.h, ot[4]);
    }
  }

  // normalize by accumulated row sums and write out
#pragma unroll
  for (int r = 0; r < 8; ++r) {
    const float inv = 1.0f / ot[4][r];
    const int m_loc = lhalf * 8 + r;
    f16* orow = O + ((size_t)(b * SEQ + qbase + m_loc)) * HID + h * HDIM;
#pragma unroll
    for (int j = 0; j < 4; ++j) orow[j * 16 + l15] = (f16)(ot[j][r] * inv);
  }
}

// ---------------------------------------------------------------------------
// launch
// ---------------------------------------------------------------------------
extern "C" void kernel_launch(void* const* d_in, const int* in_sizes, int n_in,
                              void* d_out, int out_size, void* d_ws,
                              size_t ws_size, hipStream_t stream) {
  const float* x_img = (const float*)d_in[0];
  const float* cond  = (const float*)d_in[1];
  const float* wq = (const float*)d_in[2];  const float* bq = (const float*)d_in[3];
  const float* wk = (const float*)d_in[4];  const float* bk = (const float*)d_in[5];
  const float* wv = (const float*)d_in[6];  const float* bv = (const float*)d_in[7];
  const float* wo = (const float*)d_in[8];  const float* bo = (const float*)d_in[9];
  const float* w_ada = (const float*)d_in[10]; const float* b_ada = (const float*)d_in[11];
  const float* w1 = (const float*)d_in[12]; const float* b1 = (const float*)d_in[13];
  const float* w2 = (const float*)d_in[14]; const float* b2 = (const float*)d_in[15];
  const float* ln1s = (const float*)d_in[16]; const float* ln1b = (const float*)d_in[17];
  const float* ln2s = (const float*)d_in[18]; const float* ln2b = (const float*)d_in[19];

  // workspace carve-up (~392 MiB total)
  char* ws = (char*)d_ws;
  float* c = (float*)ws;                 ws += (size_t)BATCH * ADA * 4;
  const int WSZ = HID * HID;             // 589824
  f16* Wq = (f16*)ws; ws += (size_t)WSZ * 2;
  f16* Wk = (f16*)ws; ws += (size_t)WSZ * 2;
  f16* Wv = (f16*)ws; ws += (size_t)WSZ * 2;
  f16* Wo = (f16*)ws; ws += (size_t)WSZ * 2;
  f16* W1 = (f16*)ws; ws += (size_t)WSZ * 2;
  f16* W2 = (f16*)ws; ws += (size_t)WSZ * 2;
  const size_t ACT = (size_t)NROW * HID; // 50,331,648 elems
  f16* bufH = (f16*)ws; ws += ACT * 2;   // modulated h / attn-out / h2
  f16* bufQ = (f16*)ws; ws += ACT * 2;   // Q / mlp mid
  f16* bufK = (f16*)ws; ws += ACT * 2;
  f16* bufV = (f16*)ws; ws += ACT * 2;
  float* X = (float*)d_out;              // residual stream x (f32)

  // 1. adaLN vector c = silu(cond) @ w_ada + b_ada
  adaln_kernel<<<dim3(BATCH, ADA / 256), 256, 0, stream>>>(cond, w_ada, b_ada, c);

  // 2. weights -> f16
  const int cgrid = (WSZ + 255) / 256;
  cvt_f16_kernel<<<cgrid, 256, 0, stream>>>(wq, Wq, WSZ);
  cvt_f16_kernel<<<cgrid, 256, 0, stream>>>(wk, Wk, WSZ);
  cvt_f16_kernel<<<cgrid, 256, 0, stream>>>(wv, Wv, WSZ);
  cvt_f16_kernel<<<cgrid, 256, 0, stream>>>(wo, Wo, WSZ);
  cvt_f16_kernel<<<cgrid, 256, 0, stream>>>(w1, W1, WSZ);
  cvt_f16_kernel<<<cgrid, 256, 0, stream>>>(w2, W2, WSZ);

  // 3. h = modulate(LN1(x_img), shift_msa, scale_msa)
  ln_mod_kernel<<<NROW / 8, 256, 0, stream>>>(x_img, c, 0 * HID, 1 * HID,
                                              ln1s, ln1b, bufH);

  // 4. Q/K/V projections (Q pre-scaled by 1/sqrt(d)*log2(e))
  dim3 gg(NROW / 128, HID / 128);
  gemm_wmma_kernel<3><<<gg, 256, 0, stream>>>(bufH, Wq, bq, bufQ, nullptr,
                                              nullptr, NROW, HID, HID);
  gemm_wmma_kernel<0><<<gg, 256, 0, stream>>>(bufH, Wk, bk, bufK, nullptr,
                                              nullptr, NROW, HID, HID);
  gemm_wmma_kernel<0><<<gg, 256, 0, stream>>>(bufH, Wv, bv, bufV, nullptr,
                                              nullptr, NROW, HID, HID);

  // 5. attention -> bufH (reuse)
  flash_attn_kernel<<<BATCH * HEADS * (SEQ / 128), 256, 0, stream>>>(
      bufQ, bufK, bufV, bufH);

  // 6. x = x_img + gate_msa * (attn @ wo + bo)   -> d_out (f32)
  gemm_wmma_kernel<2><<<gg, 256, 0, stream>>>(bufH, Wo, bo, X, x_img,
                                              c + 2 * HID, NROW, HID, HID);

  // 7. h2 = modulate(LN2(x), shift_mlp, scale_mlp) -> bufH
  ln_mod_kernel<<<NROW / 8, 256, 0, stream>>>(X, c, 3 * HID, 4 * HID,
                                              ln2s, ln2b, bufH);

  // 8. mid = silu(h2 @ w1 + b1) -> bufQ (reuse)
  gemm_wmma_kernel<1><<<gg, 256, 0, stream>>>(bufH, W1, b1, bufQ, nullptr,
                                              nullptr, NROW, HID, HID);

  // 9. out = x + gate_mlp * (mid @ w2 + b2)  (in-place on d_out)
  gemm_wmma_kernel<2><<<gg, 256, 0, stream>>>(bufQ, W2, b2, X, X,
                                              c + 5 * HID, NROW, HID, HID);
}
